// DynamicConvCrsMod_51402168599362
// MI455X (gfx1250) — matrix operand order repaired and checked by reading, not compiled, not run on hardware
//
#include <hip/hip_runtime.h>

// ---------------------------------------------------------------------------
// DynamicConv head, fused for MI455X (gfx1250, wave32, WMMA f16->f32).
// Never materializes param1/param2 (1.4 GB): virtual outer-product A matrices
// feed v_wmma_f32_16x16x32_f16 directly; W slabs staged through LDS.
// B-fragments are batched into distinct registers ahead of the WMMA bursts so
// ds_load latency overlaps matrix execution (partial dscnt waits).
// ---------------------------------------------------------------------------

typedef __attribute__((ext_vector_type(16))) _Float16 v16h;
typedef __attribute__((ext_vector_type(8)))  _Float16 v8h;
typedef __attribute__((ext_vector_type(8)))  float    v8f;

#define BB     7200      // BS*NQ
#define EDIM   256
#define DDYN   64
#define TWOE   512

// --- helpers ---------------------------------------------------------------

static __device__ inline v16h cat8(v8h lo, v8h hi) {
  v16h r;
#pragma unroll
  for (int i = 0; i < 8; ++i) { r[i] = lo[i]; r[i + 8] = hi[i]; }
  return r;
}

static __device__ inline v16h splat16(_Float16 s) {
  v16h r;
#pragma unroll
  for (int i = 0; i < 16; ++i) r[i] = s;
  return r;
}

static __device__ inline v8f zero8() {
  v8f z;
#pragma unroll
  for (int i = 0; i < 8; ++i) z[i] = 0.0f;
  return z;
}

// A operand (16x32 f16, M x K): lane holds row (lane&15); lanes<16 take
// K {0..7,16..23}, lanes>=16 take K {8..15,24..31}  (ISA 7.12.2 layout).
static __device__ inline v16h load_a(const _Float16* __restrict__ rowp, int c, int lh) {
  const _Float16* p = rowp + c * 32 + 8 * lh;
  v8h lo = *(const v8h*)(p);
  v8h hi = *(const v8h*)(p + 16);
  return cat8(lo, hi);
}

// B operand (32x16 f16, K x N): lane holds column (lane&15); lanes<16 take
// K {0..7,16..23}, lanes>=16 take K {8..15,24..31} (same order as A).
static __device__ inline v16h load_b(const _Float16* __restrict__ base, int col,
                                     int stride, int lh) {
  const _Float16* p = base + (size_t)col * stride + 8 * lh;
  v8h lo = *(const v8h*)(p);
  v8h hi = *(const v8h*)(p + 16);
  return cat8(lo, hi);
}

static __device__ inline v8f wmma_f16(v16h a, v16h b, v8f c) {
  return __builtin_amdgcn_wmma_f32_16x16x32_f16(false, a, false, b, (short)0, c,
                                                false, false);
}

// --- conversion kernels ----------------------------------------------------

__global__ void cvt_kernel(const float* __restrict__ s, _Float16* __restrict__ d, long n) {
  long i = (long)blockIdx.x * blockDim.x + threadIdx.x;
  if (i < n) d[i] = (_Float16)s[i];
}

// src (R x C) row-major -> dst (C x R) f16 (for bias matrices as WMMA-B feeds)
__global__ void cvtT_kernel(const float* __restrict__ s, _Float16* __restrict__ d,
                            int R, int C) {
  int i = blockIdx.x * blockDim.x + threadIdx.x;
  if (i < R * C) {
    int r = i / C, c = i % C;
    d[(size_t)c * R + r] = (_Float16)s[i];
  }
}

// --- Stage 1: f1[b,d] = sum_{k,e} feats[b,k] q[b,e] Wpre[k*64+d, e]  (+ feats@Bp)
// GEMM M=7200 K=131072 N=64.  WG = 8 waves x 32 rows = 256 rows.
// Per kk (512): 32KB W slab (rows kk*64..+64 of Wpre) staged in LDS.
__global__ __launch_bounds__(256) void stage1_kernel(
    const _Float16* __restrict__ q16, const _Float16* __restrict__ feats16,
    const _Float16* __restrict__ wpre16, const _Float16* __restrict__ bpreT16,
    float* __restrict__ f1raw) {
  __shared__ _Float16 slab[2][DDYN * EDIM];   // 2 x 32 KB
  const int tid = threadIdx.x;
  const int lane = tid & 31, wave = tid >> 5;
  const int lh = lane >> 4, ln16 = lane & 15;

  const int rowBase = blockIdx.x * 256 + wave * 32;
  int r0 = rowBase + ln16;       if (r0 >= BB) r0 = BB - 1;
  int r1 = rowBase + 16 + ln16;  if (r1 >= BB) r1 = BB - 1;

  // Preload q A-fragments for both 16-row M tiles (K=256 -> 8 chunks).
  v8h qlo[2][8], qhi[2][8];
  {
    const _Float16* p0 = q16 + (size_t)r0 * EDIM;
    const _Float16* p1 = q16 + (size_t)r1 * EDIM;
#pragma unroll
    for (int c = 0; c < 8; ++c) {
      qlo[0][c] = *(const v8h*)(p0 + c * 32 + 8 * lh);
      qhi[0][c] = *(const v8h*)(p0 + c * 32 + 16 + 8 * lh);
      qlo[1][c] = *(const v8h*)(p1 + c * 32 + 8 * lh);
      qhi[1][c] = *(const v8h*)(p1 + c * 32 + 16 + 8 * lh);
    }
  }
  const _Float16* fpr0 = feats16 + (size_t)r0 * TWOE;
  const _Float16* fpr1 = feats16 + (size_t)r1 * TWOE;

  v8f acc[2][4];
#pragma unroll
  for (int m = 0; m < 2; ++m)
#pragma unroll
    for (int n = 0; n < 4; ++n) acc[m][n] = zero8();

  // Prologue: fill buffer 0 with slab for kk=0 (contiguous 32 KB).
  {
    const v8h* src = (const v8h*)(wpre16);
    v8h* dst = (v8h*)(&slab[0][0]);
#pragma unroll
    for (int i = 0; i < 8; ++i) dst[tid + i * 256] = src[tid + i * 256];
  }

#pragma unroll 1
  for (int kk = 0; kk < TWOE; ++kk) {
    const int p = kk & 1;
    __syncthreads();   // slab[p] filled; previous readers of slab[p^1] done
    if (kk + 1 < TWOE) {
      const v8h* src = (const v8h*)(wpre16 + (size_t)(kk + 1) * DDYN * EDIM);
      v8h* dst = (v8h*)(&slab[p ^ 1][0]);
#pragma unroll
      for (int i = 0; i < 8; ++i) dst[tid + i * 256] = src[tid + i * 256];
      if (kk + 2 < TWOE)  // global_prefetch_b8 on the slab after next
        __builtin_prefetch(wpre16 + (size_t)(kk + 2) * DDYN * EDIM + tid * 64, 0, 1);
    }
    // virtual A rows: feats[b,kk] * q[b, :]
    const v16h f0 = splat16(fpr0[kk]);
    const v16h f1 = splat16(fpr1[kk]);
#pragma unroll
    for (int c = 0; c < 8; ++c) {
      // batch all 4 B fragments (8 ds_load_b128 in flight), then 8 WMMAs
      v16h b0 = load_b(&slab[p][0] + c * 32, 0 * 16 + ln16, EDIM, lh);
      v16h b1 = load_b(&slab[p][0] + c * 32, 1 * 16 + ln16, EDIM, lh);
      v16h b2 = load_b(&slab[p][0] + c * 32, 2 * 16 + ln16, EDIM, lh);
      v16h b3 = load_b(&slab[p][0] + c * 32, 3 * 16 + ln16, EDIM, lh);
      v16h a0 = cat8(qlo[0][c], qhi[0][c]) * f0;   // v_pk_mul_f16 x8
      v16h a1 = cat8(qlo[1][c], qhi[1][c]) * f1;
      acc[0][0] = wmma_f16(a0, b0, acc[0][0]);
      acc[1][0] = wmma_f16(a1, b0, acc[1][0]);
      acc[0][1] = wmma_f16(a0, b1, acc[0][1]);
      acc[1][1] = wmma_f16(a1, b1, acc[1][1]);
      acc[0][2] = wmma_f16(a0, b2, acc[0][2]);
      acc[1][2] = wmma_f16(a1, b2, acc[1][2]);
      acc[0][3] = wmma_f16(a0, b3, acc[0][3]);
      acc[1][3] = wmma_f16(a1, b3, acc[1][3]);
    }
  }

  // Fold b_pre:  f1 += feats(16x512) @ BpT(512x64);  BpT[d*512+k] K-contiguous.
#pragma unroll 1
  for (int c2 = 0; c2 < 16; ++c2) {
    v16h a0 = load_a(fpr0, c2, lh);
    v16h a1 = load_a(fpr1, c2, lh);
    v16h b0 = load_b(bpreT16 + c2 * 32, 0 * 16 + ln16, TWOE, lh);
    v16h b1 = load_b(bpreT16 + c2 * 32, 1 * 16 + ln16, TWOE, lh);
    v16h b2 = load_b(bpreT16 + c2 * 32, 2 * 16 + ln16, TWOE, lh);
    v16h b3 = load_b(bpreT16 + c2 * 32, 3 * 16 + ln16, TWOE, lh);
    acc[0][0] = wmma_f16(a0, b0, acc[0][0]);
    acc[1][0] = wmma_f16(a1, b0, acc[1][0]);
    acc[0][1] = wmma_f16(a0, b1, acc[0][1]);
    acc[1][1] = wmma_f16(a1, b1, acc[1][1]);
    acc[0][2] = wmma_f16(a0, b2, acc[0][2]);
    acc[1][2] = wmma_f16(a1, b2, acc[1][2]);
    acc[0][3] = wmma_f16(a0, b3, acc[0][3]);
    acc[1][3] = wmma_f16(a1, b3, acc[1][3]);
  }

  // C layout: vgpr v, lane -> row = base + mt*16 + v + 8*(lane>=16), col = nt*16+(lane&15)
#pragma unroll
  for (int mt = 0; mt < 2; ++mt)
#pragma unroll
    for (int nt = 0; nt < 4; ++nt)
#pragma unroll
      for (int v = 0; v < 8; ++v) {
        int row = rowBase + mt * 16 + v + 8 * lh;
        if (row < BB) f1raw[(size_t)row * DDYN + nt * 16 + ln16] = acc[mt][nt][v];
      }
}

// --- LayerNorm(D=64) + ReLU -> f16 ; one wave per row --------------------
__global__ __launch_bounds__(256) void ln64_kernel(
    const float* __restrict__ raw, const float* __restrict__ g,
    const float* __restrict__ be, _Float16* __restrict__ outh) {
  const int lane = threadIdx.x & 31, wave = threadIdx.x >> 5;
  const size_t row = (size_t)blockIdx.x * 8 + wave;
  if (row >= BB) return;
  const float2 x = *(const float2*)(raw + row * DDYN + lane * 2);
  float s = x.x + x.y, ss = x.x * x.x + x.y * x.y;
#pragma unroll
  for (int o = 16; o > 0; o >>= 1) {
    s += __shfl_xor(s, o, 32);
    ss += __shfl_xor(ss, o, 32);
  }
  const float mu = s * (1.0f / 64.0f);
  const float rs = rsqrtf(ss * (1.0f / 64.0f) - mu * mu + 1e-5f);
  const float2 gg = *(const float2*)(g + lane * 2);
  const float2 bb = *(const float2*)(be + lane * 2);
  _Float16* op = outh + row * DDYN + lane * 2;
  op[0] = (_Float16)fmaxf((x.x - mu) * rs * gg.x + bb.x, 0.0f);
  op[1] = (_Float16)fmaxf((x.y - mu) * rs * gg.y + bb.y, 0.0f);
}

// --- Stage 2: f2[b,e'] = sum_{d,e} f1r[b,d] q[b,e] Waft[d*256+e', e] (+ f1r@BaT)
// GEMM M=7200 K=16384 N=256.  WG = 8 waves x 16 rows.  64-wide e chunks in LDS.
__global__ __launch_bounds__(256) void stage2_kernel(
    const _Float16* __restrict__ q16, const _Float16* __restrict__ f1h,
    const _Float16* __restrict__ waft16, const _Float16* __restrict__ baftT16,
    float* __restrict__ f2raw) {
  __shared__ _Float16 chunk[2][EDIM * 64];   // 2 x 32 KB : rows e'(256) x 64 e
  const int tid = threadIdx.x;
  const int lane = tid & 31, wave = tid >> 5;
  const int lh = lane >> 4, ln16 = lane & 15;

  const int rowBase = blockIdx.x * 128 + wave * 16;
  int r0 = rowBase + ln16; if (r0 >= BB) r0 = BB - 1;
  const _Float16* qp = q16 + (size_t)r0 * EDIM;
  const _Float16* fp = f1h + (size_t)r0 * DDYN;

  v8h qlo[8], qhi[8];
#pragma unroll
  for (int c = 0; c < 8; ++c) {
    qlo[c] = *(const v8h*)(qp + c * 32 + 8 * lh);
    qhi[c] = *(const v8h*)(qp + c * 32 + 16 + 8 * lh);
  }
  v8f acc[16];
#pragma unroll
  for (int n = 0; n < 16; ++n) acc[n] = zero8();

  // Prologue fill: (dd=0, e0=0)
  {
    v8h* dst = (v8h*)&chunk[0][0];
#pragma unroll
    for (int i = 0; i < 8; ++i) {
      int j = tid + i * 256;
      int ep = j >> 3, c8 = j & 7;
      dst[j] = *(const v8h*)(waft16 + (size_t)ep * EDIM + c8 * 8);
    }
  }

#pragma unroll 1
  for (int kb = 0; kb < 256; ++kb) {       // dd = kb>>2 (64), e-chunk = kb&3 (4x64)
    const int p = kb & 1;
    const int dd = kb >> 2, cc = kb & 3;
    __syncthreads();
    if (kb + 1 < 256) {
      const int dd2 = (kb + 1) >> 2;
      const int e02 = ((kb + 1) & 3) * 64;
      v8h* dst = (v8h*)&chunk[p ^ 1][0];
#pragma unroll
      for (int i = 0; i < 8; ++i) {
        int j = tid + i * 256;
        int ep = j >> 3, c8 = j & 7;
        dst[j] = *(const v8h*)(waft16 + ((size_t)dd2 * EDIM + ep) * EDIM + e02 + c8 * 8);
      }
      if (kb + 2 < 256)
        __builtin_prefetch(waft16 + ((size_t)((kb + 2) >> 2) * EDIM + tid) * EDIM +
                               ((kb + 2) & 3) * 64, 0, 1);
    }
    const v16h fs = splat16(fp[dd]);       // virtual A rows: f1r[b,dd] * q[b,:]
#pragma unroll
    for (int sub = 0; sub < 2; ++sub) {
      const int c = cc * 2 + sub;
      v16h a = cat8(qlo[c], qhi[c]) * fs;
#pragma unroll
      for (int grp = 0; grp < 4; ++grp) {
        v16h b0 = load_b(&chunk[p][0] + sub * 32, (grp * 4 + 0) * 16 + ln16, 64, lh);
        v16h b1 = load_b(&chunk[p][0] + sub * 32, (grp * 4 + 1) * 16 + ln16, 64, lh);
        v16h b2 = load_b(&chunk[p][0] + sub * 32, (grp * 4 + 2) * 16 + ln16, 64, lh);
        v16h b3 = load_b(&chunk[p][0] + sub * 32, (grp * 4 + 3) * 16 + ln16, 64, lh);
        acc[grp * 4 + 0] = wmma_f16(a, b0, acc[grp * 4 + 0]);
        acc[grp * 4 + 1] = wmma_f16(a, b1, acc[grp * 4 + 1]);
        acc[grp * 4 + 2] = wmma_f16(a, b2, acc[grp * 4 + 2]);
        acc[grp * 4 + 3] = wmma_f16(a, b3, acc[grp * 4 + 3]);
      }
    }
  }

  // Fold b_aft:  f2 += f1r(16x64) @ BaT(64x256);  BaT[e'*64+d] K-contiguous.
#pragma unroll 1
  for (int c2 = 0; c2 < 2; ++c2) {
    v16h a = load_a(fp, c2, lh);
#pragma unroll
    for (int grp = 0; grp < 4; ++grp) {
      v16h b0 = load_b(baftT16 + c2 * 32, (grp * 4 + 0) * 16 + ln16, DDYN, lh);
      v16h b1 = load_b(baftT16 + c2 * 32, (grp * 4 + 1) * 16 + ln16, DDYN, lh);
      v16h b2 = load_b(baftT16 + c2 * 32, (grp * 4 + 2) * 16 + ln16, DDYN, lh);
      v16h b3 = load_b(baftT16 + c2 * 32, (grp * 4 + 3) * 16 + ln16, DDYN, lh);
      acc[grp * 4 + 0] = wmma_f16(a, b0, acc[grp * 4 + 0]);
      acc[grp * 4 + 1] = wmma_f16(a, b1, acc[grp * 4 + 1]);
      acc[grp * 4 + 2] = wmma_f16(a, b2, acc[grp * 4 + 2]);
      acc[grp * 4 + 3] = wmma_f16(a, b3, acc[grp * 4 + 3]);
    }
  }

#pragma unroll
  for (int nt = 0; nt < 16; ++nt)
#pragma unroll
    for (int v = 0; v < 8; ++v) {
      int row = rowBase + v + 8 * lh;
      if (row < BB) f2raw[(size_t)row * EDIM + nt * 16 + ln16] = acc[nt][v];
    }
}

// --- LayerNorm(E=256) + ReLU -> f16 --------------------------------------
__global__ __launch_bounds__(256) void ln256h_kernel(
    const float* __restrict__ raw, const float* __restrict__ g,
    const float* __restrict__ be, _Float16* __restrict__ outh) {
  const int lane = threadIdx.x & 31, wave = threadIdx.x >> 5;
  const size_t row = (size_t)blockIdx.x * 8 + wave;
  if (row >= BB) return;
  const float* rp = raw + row * EDIM + lane * 8;
  float x[8];
  float s = 0.f, ss = 0.f;
#pragma unroll
  for (int i = 0; i < 8; ++i) { x[i] = rp[i]; s += x[i]; ss += x[i] * x[i]; }
#pragma unroll
  for (int o = 16; o > 0; o >>= 1) {
    s += __shfl_xor(s, o, 32);
    ss += __shfl_xor(ss, o, 32);
  }
  const float mu = s * (1.0f / 256.0f);
  const float rs = rsqrtf(ss * (1.0f / 256.0f) - mu * mu + 1e-5f);
  _Float16* op = outh + row * EDIM + lane * 8;
#pragma unroll
  for (int i = 0; i < 8; ++i) {
    float y = (x[i] - mu) * rs * g[lane * 8 + i] + be[lane * 8 + i];
    op[i] = (_Float16)fmaxf(y, 0.0f);
  }
}

// --- Stage 3: out_raw = f2r @ W_out^T  (M=7200 K=256 N=256) ---------------
__global__ __launch_bounds__(256) void stage3_kernel(
    const _Float16* __restrict__ f2h, const _Float16* __restrict__ wout16,
    float* __restrict__ f3raw) {
  const int tid = threadIdx.x;
  const int lane = tid & 31, wave = tid >> 5;
  const int lh = lane >> 4, ln16 = lane & 15;
  const int rowBase = blockIdx.x * 128 + wave * 16;
  if (rowBase >= BB) return;          // no LDS/barriers: safe to exit
  int r0 = rowBase + ln16; if (r0 >= BB) r0 = BB - 1;
  const _Float16* ap = f2h + (size_t)r0 * EDIM;

  v8h alo[8], ahi[8];
#pragma unroll
  for (int c = 0; c < 8; ++c) {
    alo[c] = *(const v8h*)(ap + c * 32 + 8 * lh);
    ahi[c] = *(const v8h*)(ap + c * 32 + 16 + 8 * lh);
  }
  v8f acc[16];
#pragma unroll
  for (int n = 0; n < 16; ++n) acc[n] = zero8();

#pragma unroll 1
  for (int c = 0; c < 8; ++c) {
    v16h a = cat8(alo[c], ahi[c]);
#pragma unroll
    for (int grp = 0; grp < 4; ++grp) {
      v16h b0 = load_b(wout16 + c * 32, (grp * 4 + 0) * 16 + ln16, EDIM, lh);
      v16h b1 = load_b(wout16 + c * 32, (grp * 4 + 1) * 16 + ln16, EDIM, lh);
      v16h b2 = load_b(wout16 + c * 32, (grp * 4 + 2) * 16 + ln16, EDIM, lh);
      v16h b3 = load_b(wout16 + c * 32, (grp * 4 + 3) * 16 + ln16, EDIM, lh);
      acc[grp * 4 + 0] = wmma_f16(a, b0, acc[grp * 4 + 0]);
      acc[grp * 4 + 1] = wmma_f16(a, b1, acc[grp * 4 + 1]);
      acc[grp * 4 + 2] = wmma_f16(a, b2, acc[grp * 4 + 2]);
      acc[grp * 4 + 3] = wmma_f16(a, b3, acc[grp * 4 + 3]);
    }
  }
#pragma unroll
  for (int nt = 0; nt < 16; ++nt)
#pragma unroll
    for (int v = 0; v < 8; ++v) {
      int row = rowBase + v + 8 * lh;
      if (row < BB) f3raw[(size_t)row * EDIM + nt * 16 + ln16] = acc[nt][v];
    }
}

// --- final: (+b_out) -> LayerNorm(256) -> ReLU -> f32 out ------------------
__global__ __launch_bounds__(256) void ln256out_kernel(
    const float* __restrict__ raw, const float* __restrict__ bout,
    const float* __restrict__ g, const float* __restrict__ be,
    float* __restrict__ out) {
  const int lane = threadIdx.x & 31, wave = threadIdx.x >> 5;
  const size_t row = (size_t)blockIdx.x * 8 + wave;
  if (row >= BB) return;
  const float* rp = raw + row * EDIM + lane * 8;
  float x[8];
  float s = 0.f, ss = 0.f;
#pragma unroll
  for (int i = 0; i < 8; ++i) {
    x[i] = rp[i] + bout[lane * 8 + i];
    s += x[i]; ss += x[i] * x[i];
  }
#pragma unroll
  for (int o = 16; o > 0; o >>= 1) {
    s += __shfl_xor(s, o, 32);
    ss += __shfl_xor(ss, o, 32);
  }
  const float mu = s * (1.0f / 256.0f);
  const float rs = rsqrtf(ss * (1.0f / 256.0f) - mu * mu + 1e-5f);
  float* op = out + row * EDIM + lane * 8;
#pragma unroll
  for (int i = 0; i < 8; ++i) {
    float y = (x[i] - mu) * rs * g[lane * 8 + i] + be[lane * 8 + i];
    op[i] = fmaxf(y, 0.0f);
  }
}

// ---------------------------------------------------------------------------

extern "C" void kernel_launch(void* const* d_in, const int* in_sizes, int n_in,
                              void* d_out, int out_size, void* d_ws, size_t ws_size,
                              hipStream_t stream) {
  (void)in_sizes; (void)n_in; (void)out_size; (void)ws_size;
  const float* query = (const float*)d_in[0];
  const float* feats = (const float*)d_in[1];
  const float* W_pre = (const float*)d_in[2];
  const float* b_pre = (const float*)d_in[3];
  const float* W_aft = (const float*)d_in[4];
  const float* b_aft = (const float*)d_in[5];
  const float* W_out = (const float*)d_in[6];
  const float* b_out = (const float*)d_in[7];
  const float* g1 = (const float*)d_in[8];
  const float* be1 = (const float*)d_in[9];
  const float* g2 = (const float*)d_in[10];
  const float* be2 = (const float*)d_in[11];
  const float* g3 = (const float*)d_in[12];
  const float* be3 = (const float*)d_in[13];

  char* w = (char*)d_ws;
  _Float16* wpre16  = (_Float16*)(w + 0);                    // 16,777,216 B
  _Float16* waft16  = (_Float16*)(w + 16777216);             //  8,388,608 B
  _Float16* wout16  = (_Float16*)(w + 25165824);             //    131,072 B
  _Float16* q16     = (_Float16*)(w + 25296896);             //  3,686,400 B
  _Float16* feats16 = (_Float16*)(w + 28983296);             //  7,372,800 B
  _Float16* bpreT16 = (_Float16*)(w + 36356096);             //     65,536 B
  _Float16* baftT16 = (_Float16*)(w + 36421632);             //     32,768 B
  float*    f1raw   = (float*)   (w + 36454400);             //  1,843,200 B
  _Float16* f1h     = (_Float16*)(w + 38297600);             //    921,600 B
  float*    f2raw   = (float*)   (w + 39219200);             //  7,372,800 B
  _Float16* f2h     = (_Float16*)(w + 46592000);             //  3,686,400 B
  float*    f3raw   = (float*)   (w + 50278400);             //  7,372,800 B

  auto gs = [](long n) { return (unsigned)((n + 255) / 256); };

  cvt_kernel<<<gs(7200L * 256), 256, 0, stream>>>(query, q16, 7200L * 256);
  cvt_kernel<<<gs(7200L * 512), 256, 0, stream>>>(feats, feats16, 7200L * 512);
  cvt_kernel<<<gs(32768L * 256), 256, 0, stream>>>(W_pre, wpre16, 32768L * 256);
  cvt_kernel<<<gs(16384L * 256), 256, 0, stream>>>(W_aft, waft16, 16384L * 256);
  cvt_kernel<<<gs(256L * 256), 256, 0, stream>>>(W_out, wout16, 256L * 256);
  cvtT_kernel<<<gs(512 * 64), 256, 0, stream>>>(b_pre, bpreT16, 512, 64);   // (512,64)->(64,512)
  cvtT_kernel<<<gs(64 * 256), 256, 0, stream>>>(b_aft, baftT16, 64, 256);   // (64,256)->(256,64)

  stage1_kernel<<<29, 256, 0, stream>>>(q16, feats16, wpre16, bpreT16, f1raw);
  ln64_kernel<<<900, 256, 0, stream>>>(f1raw, g1, be1, f1h);
  stage2_kernel<<<57, 256, 0, stream>>>(q16, f1h, waft16, baftT16, f2raw);
  ln256h_kernel<<<900, 256, 0, stream>>>(f2raw, g2, be2, f2h);
  stage3_kernel<<<57, 256, 0, stream>>>(f2h, wout16, f3raw);
  ln256out_kernel<<<900, 256, 0, stream>>>(f3raw, b_out, g3, be3, (float*)d_out);
}
